// MeanPoolNet_46694884442560
// MI455X (gfx1250) — matrix operand order repaired and checked by reading, not compiled, not run on hardware
//
#include <hip/hip_runtime.h>
#include <hip/hip_bf16.h>

#define HID   192
#define NNODE 1024
#define NB    16
#define FDIM  128
#define MROWS (NB * NNODE)
#define KC    128   // K-chunk staged in LDS for the propagation GEMM

typedef __attribute__((ext_vector_type(16))) __bf16 bf16x16;
typedef __attribute__((ext_vector_type(8)))  __bf16 bf16x8;
typedef __attribute__((ext_vector_type(8)))  float  f32x8;
typedef int i32x4 __attribute__((vector_size(16)));

#if __has_builtin(__builtin_amdgcn_global_load_async_to_lds_b128)
#define HAVE_ASYNC_LDS 1
#else
#define HAVE_ASYNC_LDS 0
#endif

static __device__ __forceinline__ f32x8 wmma_bf16(bf16x16 a, bf16x16 b, f32x8 c) {
  // v_wmma_f32_16x16x32_bf16: D = A x B + C, f32 accum
  return __builtin_amdgcn_wmma_f32_16x16x32_bf16(false, a, false, b, (short)0, c, false, false);
}

// 16-byte global -> LDS copy; async (ASYNCcnt) when the gfx1250 builtin exists
static __device__ __forceinline__ void copy16_g2l(const __bf16* g, __bf16* l) {
#if HAVE_ASYNC_LDS
  __builtin_amdgcn_global_load_async_to_lds_b128(
      (__attribute__((address_space(1))) i32x4*)(uintptr_t)g,
      (__attribute__((address_space(3))) i32x4*)(unsigned int)(uintptr_t)l,
      0, 0);
#else
  *(bf16x8*)l = *(const bf16x8*)g;
#endif
}

static __device__ __forceinline__ void wait_async_copies() {
#if HAVE_ASYNC_LDS
#if __has_builtin(__builtin_amdgcn_s_wait_asynccnt)
  __builtin_amdgcn_s_wait_asynccnt(0);
#else
  asm volatile("s_wait_asynccnt 0" ::: "memory");
#endif
#endif
  // sync fallback: visibility handled by __syncthreads (DScnt)
}

// ---------------------------------------------------------------------------
// Transpose f32 [K,N] -> bf16 [N,K]   (weights; so B fragments are contiguous)
// ---------------------------------------------------------------------------
__global__ void k_transpose_bf16(const float* __restrict__ In, __bf16* __restrict__ Out,
                                 int K, int N) {
  int idx = blockIdx.x * 256 + threadIdx.x;
  if (idx < K * N) {
    int k = idx / N, n = idx % N;
    Out[(size_t)n * K + k] = (__bf16)In[idx];
  }
}

// ---------------------------------------------------------------------------
// deg -> dis = deg^-1/2  (self loops guarantee deg >= 1)
// ---------------------------------------------------------------------------
__global__ void k_deg(const float* __restrict__ adj, float* __restrict__ dis) {
  int b = blockIdx.y, i = blockIdx.x;
  const float* row = adj + ((size_t)b * NNODE + i) * NNODE;
  __shared__ float red[256];
  float c = 0.f;
  for (int j = threadIdx.x; j < NNODE; j += 256) c += (row[j] != 0.f) ? 1.f : 0.f;
  red[threadIdx.x] = c;
  __syncthreads();
  for (int st = 128; st > 0; st >>= 1) {
    if (threadIdx.x < st) red[threadIdx.x] += red[threadIdx.x + st];
    __syncthreads();
  }
  if (threadIdx.x == 0) {
    float deg = red[0] + ((row[i] == 0.f) ? 1.f : 0.f);  // explicit self loop
    dis[b * NNODE + i] = rsqrtf(deg);
  }
}

// ---------------------------------------------------------------------------
// A_hat[b,i,j] = dis_i * dis_j * ((adj!=0)||(i==j))  -> bf16 (fits in L2)
// ---------------------------------------------------------------------------
__global__ void k_build_ahat(const float* __restrict__ adj, const float* __restrict__ dis,
                             __bf16* __restrict__ Ahat) {
  int b = blockIdx.z, i = blockIdx.y;
  int j = blockIdx.x * 256 + threadIdx.x;
  size_t base = ((size_t)b * NNODE + i) * NNODE;
  float di = dis[b * NNODE + i];
  float a  = adj[base + j];
  float v  = (a != 0.f || j == i) ? di * dis[b * NNODE + j] : 0.f;
  Ahat[base + j] = (__bf16)v;
}

// ---------------------------------------------------------------------------
// Per-feature BN stats over M rows; emit fused scale/shift: y = x*s + t
// ---------------------------------------------------------------------------
__global__ void k_stats(const float* __restrict__ X, int M, int F,
                        const float* __restrict__ g, const float* __restrict__ bta,
                        float* __restrict__ s, float* __restrict__ t) {
  int f = blockIdx.x;
  __shared__ float ssum[256], ssq[256];
  float a = 0.f, q = 0.f;
  for (int i = threadIdx.x; i < M; i += 256) {
    float v = X[(size_t)i * F + f];
    a += v; q += v * v;
  }
  ssum[threadIdx.x] = a; ssq[threadIdx.x] = q;
  __syncthreads();
  for (int st = 128; st > 0; st >>= 1) {
    if (threadIdx.x < st) {
      ssum[threadIdx.x] += ssum[threadIdx.x + st];
      ssq[threadIdx.x]  += ssq[threadIdx.x + st];
    }
    __syncthreads();
  }
  if (threadIdx.x == 0) {
    float m   = ssum[0] / (float)M;
    float var = ssq[0] / (float)M - m * m;            // biased var (matches jnp.var)
    float sc  = g[f] * rsqrtf(var + 1e-5f);
    s[f] = sc;
    t[f] = bta[f] - m * sc;
  }
}

// ---------------------------------------------------------------------------
// WMMA GEMM with BN fused on A-load:  C = BN(X) @ W
//  TOUT=false: Out = relu(C) f32, row-major [M,HID]   (feature transform)
//  TOUT=true : OutT[b][n][m%1024] = bf16(C)           (conv pre-transform, transposed
//                                                      so it is a contiguous B operand)
// ---------------------------------------------------------------------------
template <int K, bool TOUT>
__global__ __launch_bounds__(256)
void k_gemm_bn(const float* __restrict__ X, const __bf16* __restrict__ BT,
               float* __restrict__ OutF, __bf16* __restrict__ OutT,
               const float* __restrict__ s, const float* __restrict__ t) {
  const int lane = threadIdx.x & 31;
  const int wave = threadIdx.x >> 5;
  const int wm = wave & 3, wn = wave >> 2;           // 4x2 wave tiles -> 64x32 block
  const int hf = lane >> 4, l15 = lane & 15;
  const int mtile = blockIdx.x * 64 + wm * 16;
  const int ntile = blockIdx.y * 32 + wn * 16;
  const int row = mtile + l15;
  const int col = ntile + l15;
  const float*  xr = X  + (size_t)row * K;
  const __bf16* br = BT + (size_t)col * K;

  f32x8 acc = {};
  for (int k = 0; k < K; k += 32) {
    int ka = k + hf * 8;
    f32x8 x0 = *(const f32x8*)(xr + ka);
    f32x8 x1 = *(const f32x8*)(xr + ka + 16);
    f32x8 s0 = *(const f32x8*)(s + ka);
    f32x8 s1 = *(const f32x8*)(s + ka + 16);
    f32x8 t0 = *(const f32x8*)(t + ka);
    f32x8 t1 = *(const f32x8*)(t + ka + 16);
    union { bf16x16 v; bf16x8 h[2]; } ua;
    ua.h[0] = __builtin_convertvector(x0 * s0 + t0, bf16x8);
    ua.h[1] = __builtin_convertvector(x1 * s1 + t1, bf16x8);
    union { bf16x16 v; bf16x8 h[2]; } ub;
    ub.h[0] = *(const bf16x8*)(br + k + hf * 8);
    ub.h[1] = *(const bf16x8*)(br + k + hf * 8 + 16);
    bf16x16 bfrag = ub.v;
    acc = wmma_bf16(ua.v, bfrag, acc);
  }
#pragma unroll
  for (int r = 0; r < 8; ++r) {
    int m = mtile + hf * 8 + r;
    float v = acc[r];
    if constexpr (TOUT) {
      int b = m >> 10, mm = m & 1023;
      OutT[((size_t)b * HID + col) * NNODE + mm] = (__bf16)v;
    } else {
      OutF[(size_t)m * HID + col] = fmaxf(v, 0.f);
    }
  }
}

// ---------------------------------------------------------------------------
// Batched propagation GEMM:  H = relu(A_hat_b @ t_b + bias)
// A = Ahat bf16 [1024,1024] row-major, streamed from global (L2-resident).
// B = tT bf16 [HID,1024]: 32x128 panels double-buffered in LDS via async copy,
//     consumed with ds_load_b128 -> removes the 4x redundant per-wave B reads.
// ---------------------------------------------------------------------------
__global__ __launch_bounds__(256)
void k_gemm_ahat(const __bf16* __restrict__ Ahat, const __bf16* __restrict__ tT,
                 const float* __restrict__ bias, float* __restrict__ Out) {
  __shared__ __bf16 shB[2][32 * KC];   // 2 x 8KB double buffer

  const int lane = threadIdx.x & 31;
  const int wave = threadIdx.x >> 5;
  const int wm = wave & 3, wn = wave >> 2;
  const int hf = lane >> 4, l15 = lane & 15;
  const int b = blockIdx.z;
  const int mtile = blockIdx.x * 64 + wm * 16;
  const int ntile0 = blockIdx.y * 32;
  const int row = mtile + l15;
  const __bf16* ar = Ahat + ((size_t)b * NNODE + row) * NNODE;

  // panel copy mapping: 32 cols x KC elems; 256 threads x 2 x 16B
  const int cc = threadIdx.x >> 3;          // 0..31 local col
  const int o8 = (threadIdx.x & 7) * 8;     // 0..56 elem offset
  const __bf16* srcBase = tT + ((size_t)b * HID + ntile0 + cc) * NNODE + o8;
  // B fragment source in LDS for this wave
  const __bf16* bfragBase = &shB[0][(wn * 16 + l15) * KC + hf * 8];
  const int bufStride = 32 * KC;

  // prologue: stage panel 0
  {
    const __bf16* src = srcBase;
    __bf16* dst = &shB[0][cc * KC + o8];
    copy16_g2l(src, dst);
    copy16_g2l(src + 64, dst + 64);
  }

  f32x8 acc = {};
  int buf = 0;
  for (int chunk = 0; chunk < NNODE / KC; ++chunk) {
    wait_async_copies();
    __syncthreads();                         // panel `chunk` visible to all waves
    if (chunk + 1 < NNODE / KC) {            // stage next panel into other buffer
      const __bf16* src = srcBase + (chunk + 1) * KC;
      __bf16* dst = &shB[0][(buf ^ 1) * bufStride + cc * KC + o8];
      copy16_g2l(src, dst);
      copy16_g2l(src + 64, dst + 64);
      __builtin_prefetch(ar + (chunk + 1) * KC, 0, 1);   // global_prefetch next A chunk
    }
    const int kc = chunk * KC;
#pragma unroll
    for (int kk = 0; kk < KC; kk += 32) {
      int ka = kc + kk + hf * 8;
      union { bf16x16 v; bf16x8 h[2]; } ua;
      ua.h[0] = *(const bf16x8*)(ar + ka);
      ua.h[1] = *(const bf16x8*)(ar + ka + 16);
      union { bf16x16 v; bf16x8 h[2]; } ub;
      ub.h[0] = *(const bf16x8*)(bfragBase + buf * bufStride + kk);
      ub.h[1] = *(const bf16x8*)(bfragBase + buf * bufStride + kk + 16);
      bf16x16 bfrag = ub.v;
      acc = wmma_bf16(ua.v, bfrag, acc);
    }
    buf ^= 1;
  }

  const int col = ntile0 + wn * 16 + l15;
  float bb = bias[col];
#pragma unroll
  for (int r = 0; r < 8; ++r) {
    int m = b * NNODE + mtile + hf * 8 + r;
    Out[(size_t)m * HID + col] = fmaxf(acc[r] + bb, 0.f);
  }
}

// ---------------------------------------------------------------------------
// global mean pool over nodes
// ---------------------------------------------------------------------------
__global__ void k_pool(const float* __restrict__ H, float* __restrict__ G) {
  int b = blockIdx.x, h = threadIdx.x;  // 192 threads
  float s = 0.f;
  for (int n = 0; n < NNODE; ++n) s += H[((size_t)b * NNODE + n) * HID + h];
  G[b * HID + h] = s * (1.f / NNODE);
}

// ---------------------------------------------------------------------------
// Fused head: BN -> relu(lin) -> BN -> cls -> log_softmax   (16x192, single block)
// ---------------------------------------------------------------------------
__global__ void k_head(const float* __restrict__ G,
                       const float* __restrict__ fcg, const float* __restrict__ fcb,
                       const float* __restrict__ linW, const float* __restrict__ linb,
                       const float* __restrict__ hidg, const float* __restrict__ hidb,
                       const float* __restrict__ clsW, const float* __restrict__ clsb,
                       float* __restrict__ out) {
  __shared__ float g1[NB * HID], g2[NB * HID], logit[NB * 10], lse[NB];
  int h = threadIdx.x;  // 0..191
  float m = 0.f, v = 0.f;
  for (int b = 0; b < NB; ++b) { float x = G[b * HID + h]; m += x; v += x * x; }
  m *= (1.f / NB); v = v * (1.f / NB) - m * m;
  float sc = fcg[h] * rsqrtf(v + 1e-5f), sh = fcb[h] - m * sc;
  for (int b = 0; b < NB; ++b) g1[b * HID + h] = G[b * HID + h] * sc + sh;
  __syncthreads();
  for (int b = 0; b < NB; ++b) {
    float a = linb[h];
    for (int j = 0; j < HID; ++j) a += g1[b * HID + j] * linW[j * HID + h];
    g2[b * HID + h] = fmaxf(a, 0.f);
  }
  __syncthreads();
  m = 0.f; v = 0.f;
  for (int b = 0; b < NB; ++b) { float x = g2[b * HID + h]; m += x; v += x * x; }
  m *= (1.f / NB); v = v * (1.f / NB) - m * m;
  sc = hidg[h] * rsqrtf(v + 1e-5f); sh = hidb[h] - m * sc;
  for (int b = 0; b < NB; ++b) g1[b * HID + h] = g2[b * HID + h] * sc + sh;
  __syncthreads();
  if (h < NB * 10) {
    int b = h / 10, c = h % 10;
    float a = clsb[c];
    for (int j = 0; j < HID; ++j) a += g1[b * HID + j] * clsW[j * 10 + c];
    logit[h] = a;
  }
  __syncthreads();
  if (h < NB) {
    float mx = -3.402823e38f;
    for (int c = 0; c < 10; ++c) mx = fmaxf(mx, logit[h * 10 + c]);
    float se = 0.f;
    for (int c = 0; c < 10; ++c) se += __expf(logit[h * 10 + c] - mx);
    lse[h] = mx + __logf(se);
  }
  __syncthreads();
  if (h < NB * 10) out[h] = logit[h] - lse[h / 10];
}

// ---------------------------------------------------------------------------
extern "C" void kernel_launch(void* const* d_in, const int* in_sizes, int n_in,
                              void* d_out, int out_size, void* d_ws, size_t ws_size,
                              hipStream_t stream) {
  const float* x        = (const float*)d_in[0];
  const float* adj      = (const float*)d_in[1];
  const float* bnfg     = (const float*)d_in[2];
  const float* bnfb     = (const float*)d_in[3];
  const float* W_feat   = (const float*)d_in[4];
  const float* bcg      = (const float*)d_in[5];
  const float* bcb      = (const float*)d_in[6];
  const float* convs_W  = (const float*)d_in[7];
  const float* convs_b  = (const float*)d_in[8];
  const float* bn_fc_g  = (const float*)d_in[9];
  const float* bn_fc_b  = (const float*)d_in[10];
  const float* lin_W    = (const float*)d_in[11];
  const float* lin_b    = (const float*)d_in[12];
  const float* bn_hid_g = (const float*)d_in[13];
  const float* bn_hid_b = (const float*)d_in[14];
  const float* cls_W    = (const float*)d_in[15];
  const float* cls_b    = (const float*)d_in[16];

  char* ws = (char*)d_ws;
  size_t off = 0;
  auto take = [&](size_t bytes) -> char* {
    char* p = ws + off;
    off += (bytes + 255) & ~(size_t)255;
    return p;
  };
  __bf16* Ahat = (__bf16*)take((size_t)NB * NNODE * NNODE * 2);   // 32 MB
  float*  H    = (float*) take((size_t)MROWS * HID * 4);          // 12 MB
  __bf16* tT   = (__bf16*)take((size_t)NB * HID * NNODE * 2);     //  6 MB
  float*  dis  = (float*) take((size_t)NB * NNODE * 4);
  __bf16* WfT  = (__bf16*)take((size_t)HID * FDIM * 2);
  __bf16* WcT  = (__bf16*)take((size_t)3 * HID * HID * 2);
  float*  sbuf = (float*) take(HID * 4);
  float*  tbuf = (float*) take(HID * 4);
  float*  G    = (float*) take((size_t)NB * HID * 4);

  // weights -> bf16 transposed
  k_transpose_bf16<<<dim3((FDIM * HID + 255) / 256), 256, 0, stream>>>(W_feat, WfT, FDIM, HID);
  for (int i = 0; i < 3; ++i)
    k_transpose_bf16<<<dim3((HID * HID + 255) / 256), 256, 0, stream>>>(
        convs_W + (size_t)i * HID * HID, WcT + (size_t)i * HID * HID, HID, HID);

  // normalized adjacency (bf16, L2-resident)
  k_deg<<<dim3(NNODE, NB), 256, 0, stream>>>(adj, dis);
  k_build_ahat<<<dim3(NNODE / 256, NNODE, NB), 256, 0, stream>>>(adj, dis, Ahat);

  // feature transform: H = relu(BN(x) @ W_feat)
  k_stats<<<FDIM, 256, 0, stream>>>(x, MROWS, FDIM, bnfg, bnfb, sbuf, tbuf);
  k_gemm_bn<FDIM, false><<<dim3(MROWS / 64, HID / 32), 256, 0, stream>>>(
      x, WfT, H, nullptr, sbuf, tbuf);

  // 3 x GCN conv
  for (int i = 0; i < 3; ++i) {
    k_stats<<<HID, 256, 0, stream>>>(H, MROWS, HID, bcg + i * HID, bcb + i * HID, sbuf, tbuf);
    k_gemm_bn<HID, true><<<dim3(MROWS / 64, HID / 32), 256, 0, stream>>>(
        H, WcT + (size_t)i * HID * HID, nullptr, tT, sbuf, tbuf);
    k_gemm_ahat<<<dim3(NNODE / 64, HID / 32, NB), 256, 0, stream>>>(
        Ahat, tT, convs_b + i * HID, H);
  }

  // pool + head
  k_pool<<<NB, HID, 0, stream>>>(H, G);
  k_head<<<1, HID, 0, stream>>>(G, bn_fc_g, bn_fc_b, lin_W, lin_b,
                                bn_hid_g, bn_hid_b, cls_W, cls_b, (float*)d_out);
}
